// MultimodalSparseDeformableTransformer_13872744366700
// MI455X (gfx1250) — matrix operand order repaired and checked by reading, not compiled, not run on hardware
//
#include <hip/hip_runtime.h>
#include <math.h>

#define BB 32
#define NQL 300
#define DM 256
#define NH 8
#define DHD 32
#define NLV 4
#define NPP 4
#define DFF 1024
#define TVID 7680
#define TAUD 2880
#define BQ (BB*NQL)            // 9600 rows
#define ATT_SCALE 0.17677669529663689f

typedef __attribute__((ext_vector_type(16))) __bf16 v16bf;
typedef __attribute__((ext_vector_type(2)))  __bf16 v2bf;
typedef __attribute__((ext_vector_type(8)))  float  v8f;

union FragBF {
  v16bf v;
  uint4 q[2];
  unsigned short u[16];
};

__device__ inline unsigned short f2bf(float f) {
  return __builtin_bit_cast(unsigned short, (__bf16)f);   // fptrunc, RNE
}
__device__ inline float bf2f(unsigned short s) {
  return __uint_as_float(((unsigned int)s) << 16);
}
__device__ inline unsigned int pack2bf(float lo, float hi) {
#if __has_builtin(__builtin_amdgcn_cvt_pk_bf16_f32)
  v2bf r = __builtin_amdgcn_cvt_pk_bf16_f32(lo, hi);
  return __builtin_bit_cast(unsigned int, r);
#else
  return (unsigned int)f2bf(lo) | ((unsigned int)f2bf(hi) << 16);
#endif
}
__device__ inline float wave_reduce_add(float v) {
  #pragma unroll
  for (int o = 16; o > 0; o >>= 1) v += __shfl_xor(v, o, 32);
  return v;
}

// ---------------------------------------------------------------------------
// Generic GEMM: C[M,N] = A[M,K] * W[K,N] + bias, bf16 WMMA, f32 accumulate.
// Block = 256 threads = 8 waves; tile 64(M) x 64(N); waves arranged 4(M)x2(N),
// each wave computes two 16x16 WMMA tiles sharing one A fragment.
// Double-buffered LDS: next tile's float4 global loads issued before the
// current tile's WMMAs, packed (v_cvt_pk_bf16_f32) + stored afterwards;
// single barrier per K-step. Fragments load as ds_load_b128.
// Requires M%64==0, N%64==0, K%32==0 (true for every GEMM here).
// ---------------------------------------------------------------------------
template<int RELU, int STORE_BF16, int USE_MASK>
__global__ __launch_bounds__(256) void gemm_wmma(
    const float* __restrict__ A, const float* __restrict__ W,
    const float* __restrict__ bias,
    float* __restrict__ Cf, unsigned short* __restrict__ Cb,
    const unsigned char* __restrict__ rowmask,
    int M, int N, int K)
{
  __shared__ unsigned short Als[2][64][40];   // [buf][m][k], 80B row stride
  __shared__ unsigned short Bt [2][64][40];   // [buf][n][k] transposed W tile
  const int tid   = threadIdx.x;
  const int lane  = tid & 31;
  const int wave  = tid >> 5;
  const int m0    = blockIdx.y * 64;
  const int n0    = blockIdx.x * 64;
  const int ms    = (wave & 3) * 16;
  const int ns    = (wave >> 2) * 32;
  const int lrow  = lane & 15;
  const int lhalf = lane >> 4;

  // staging coordinates
  const int a_r0 = tid >> 3;                // A row (0..31), +32 for second
  const int a_c4 = (tid & 7) * 4;           // A col group
  const int w_n4 = (tid & 15) * 4;          // W: 4 consecutive n
  const int w_kp = (tid >> 4) * 2;          // W: 2 consecutive k

  const float* aptr0 = &A[(size_t)(m0 + a_r0)      * K + a_c4];
  const float* aptr1 = &A[(size_t)(m0 + a_r0 + 32) * K + a_c4];
  const float* wptr0 = &W[(size_t)(w_kp)     * N + n0 + w_n4];
  const float* wptr1 = &W[(size_t)(w_kp + 1) * N + n0 + w_n4];

  auto stage = [&](int buf, const float4& av0, const float4& av1,
                   const float4& w0, const float4& w1) {
    uint2 p0; p0.x = pack2bf(av0.x, av0.y); p0.y = pack2bf(av0.z, av0.w);
    uint2 p1; p1.x = pack2bf(av1.x, av1.y); p1.y = pack2bf(av1.z, av1.w);
    *(uint2*)&Als[buf][a_r0][a_c4]      = p0;
    *(uint2*)&Als[buf][a_r0 + 32][a_c4] = p1;
    *(unsigned int*)&Bt[buf][w_n4 + 0][w_kp] = pack2bf(w0.x, w1.x);
    *(unsigned int*)&Bt[buf][w_n4 + 1][w_kp] = pack2bf(w0.y, w1.y);
    *(unsigned int*)&Bt[buf][w_n4 + 2][w_kp] = pack2bf(w0.z, w1.z);
    *(unsigned int*)&Bt[buf][w_n4 + 3][w_kp] = pack2bf(w0.w, w1.w);
  };

  v8f acc0 = {0.f,0.f,0.f,0.f,0.f,0.f,0.f,0.f};
  v8f acc1 = {0.f,0.f,0.f,0.f,0.f,0.f,0.f,0.f};

  // Prologue: stage tile k0=0 into buffer 0
  {
    float4 av0 = *(const float4*)aptr0;
    float4 av1 = *(const float4*)aptr1;
    float4 w0  = *(const float4*)wptr0;
    float4 w1  = *(const float4*)wptr1;
    stage(0, av0, av1, w0, w1);
  }
  __syncthreads();

  int cur = 0;
  for (int k0 = 32; k0 < K; k0 += 32) {
    // Issue next tile's global loads first (latency hidden behind WMMAs)
    float4 av0 = *(const float4*)(aptr0 + k0);
    float4 av1 = *(const float4*)(aptr1 + k0);
    float4 w0  = *(const float4*)(wptr0 + (size_t)k0 * N);
    float4 w1  = *(const float4*)(wptr1 + (size_t)k0 * N);
    if (k0 + 32 < K) {
      __builtin_prefetch(aptr0 + k0 + 32, 0, 1);
      __builtin_prefetch(wptr0 + (size_t)(k0 + 32) * N, 0, 1);
    }

    FragBF fa, fb0, fb1;
    fa.q[0]  = *(const uint4*)&Als[cur][ms + lrow][lhalf * 16];
    fa.q[1]  = *(const uint4*)&Als[cur][ms + lrow][lhalf * 16 + 8];
    fb0.q[0] = *(const uint4*)&Bt[cur][ns + lrow][lhalf * 16];
    fb0.q[1] = *(const uint4*)&Bt[cur][ns + lrow][lhalf * 16 + 8];
    fb1.q[0] = *(const uint4*)&Bt[cur][ns + 16 + lrow][lhalf * 16];
    fb1.q[1] = *(const uint4*)&Bt[cur][ns + 16 + lrow][lhalf * 16 + 8];
    acc0 = __builtin_amdgcn_wmma_f32_16x16x32_bf16(false, fa.v, false, fb0.v,
                                                   (short)0, acc0, false, false);
    acc1 = __builtin_amdgcn_wmma_f32_16x16x32_bf16(false, fa.v, false, fb1.v,
                                                   (short)0, acc1, false, false);

    stage(cur ^ 1, av0, av1, w0, w1);
    __syncthreads();
    cur ^= 1;
  }

  // Final tile
  {
    FragBF fa, fb0, fb1;
    fa.q[0]  = *(const uint4*)&Als[cur][ms + lrow][lhalf * 16];
    fa.q[1]  = *(const uint4*)&Als[cur][ms + lrow][lhalf * 16 + 8];
    fb0.q[0] = *(const uint4*)&Bt[cur][ns + lrow][lhalf * 16];
    fb0.q[1] = *(const uint4*)&Bt[cur][ns + lrow][lhalf * 16 + 8];
    fb1.q[0] = *(const uint4*)&Bt[cur][ns + 16 + lrow][lhalf * 16];
    fb1.q[1] = *(const uint4*)&Bt[cur][ns + 16 + lrow][lhalf * 16 + 8];
    acc0 = __builtin_amdgcn_wmma_f32_16x16x32_bf16(false, fa.v, false, fb0.v,
                                                   (short)0, acc0, false, false);
    acc1 = __builtin_amdgcn_wmma_f32_16x16x32_bf16(false, fa.v, false, fb1.v,
                                                   (short)0, acc1, false, false);
  }

  const int gcol0 = n0 + ns + lrow;
  const float b0 = bias[gcol0];
  const float b1 = bias[gcol0 + 16];
  #pragma unroll
  for (int r = 0; r < 8; ++r) {
    const int grow = m0 + ms + r + 8 * lhalf;
    float v0 = acc0[r] + b0;
    float v1 = acc1[r] + b1;
    if (RELU) { v0 = fmaxf(v0, 0.f); v1 = fmaxf(v1, 0.f); }
    if (USE_MASK) {
      if (rowmask[grow]) { v0 = 0.f; v1 = 0.f; }
    }
    const size_t off = (size_t)grow * N + gcol0;
    if (STORE_BF16) {
      Cb[off]      = f2bf(v0);
      Cb[off + 16] = f2bf(v1);
    } else {
      Cf[off]      = v0;
      Cf[off + 16] = v1;
    }
  }
}

// ---------------------------------------------------------------------------
// Elementwise add
// ---------------------------------------------------------------------------
__global__ void add_kernel(const float* __restrict__ a, const float* __restrict__ b,
                           float* __restrict__ y, int n) {
  int i = blockIdx.x * 256 + threadIdx.x;
  if (i < n) y[i] = a[i] + b[i];
}

// ---------------------------------------------------------------------------
// MHA with online softmax. One block per (b, h, query-half-of-150).
// K/V streamed through LDS in 128-row chunks; float4 staging and compute
// (row stride 36 floats keeps 16B alignment).
// ---------------------------------------------------------------------------
__global__ __launch_bounds__(256) void attn_kernel(
    const float* __restrict__ Qp, const float* __restrict__ Kp,
    const float* __restrict__ Vp, const unsigned char* __restrict__ qmask,
    float* __restrict__ ctx)
{
  __shared__ float Kls[128][36];
  __shared__ float Vls[128][36];
  const int tid  = threadIdx.x;
  const int bh   = blockIdx.x >> 1;
  const int half = blockIdx.x & 1;
  const int b = bh >> 3, h = bh & 7;
  const int q = half * 150 + tid;
  const bool active = (tid < 150);

  float Qreg[32], o[32];
  float m = -1e30f, l = 0.f;
  #pragma unroll
  for (int d = 0; d < 32; ++d) { Qreg[d] = 0.f; o[d] = 0.f; }
  if (active) {
    const float4* qv = (const float4*)&Qp[((size_t)(b * NQL + q)) * DM + h * DHD];
    #pragma unroll
    for (int d4 = 0; d4 < 8; ++d4) {
      float4 t = qv[d4];
      Qreg[4*d4+0] = t.x; Qreg[4*d4+1] = t.y; Qreg[4*d4+2] = t.z; Qreg[4*d4+3] = t.w;
    }
  }

  for (int c0 = 0; c0 < NQL; c0 += 128) {
    int cn = NQL - c0; if (cn > 128) cn = 128;
    for (int idx = tid; idx < 128 * 8; idx += 256) {
      int r = idx >> 3, d4 = (idx & 7) * 4;
      if (r < cn) {
        size_t kb = ((size_t)(b * NQL + c0 + r)) * DM + h * DHD + d4;
        *(float4*)&Kls[r][d4] = *(const float4*)&Kp[kb];
        *(float4*)&Vls[r][d4] = *(const float4*)&Vp[kb];
      }
    }
    __syncthreads();
    if (active) {
      for (int r = 0; r < cn; ++r) {
        const float4* krow = (const float4*)&Kls[r][0];
        float s = 0.f;
        #pragma unroll
        for (int d4 = 0; d4 < 8; ++d4) {
          float4 kv = krow[d4];
          s += Qreg[4*d4 + 0] * kv.x + Qreg[4*d4 + 1] * kv.y
             + Qreg[4*d4 + 2] * kv.z + Qreg[4*d4 + 3] * kv.w;
        }
        s *= ATT_SCALE;
        if (!qmask[b * NQL + c0 + r]) s = -1e9f;
        float mn = fmaxf(m, s);
        float corr = __expf(m - mn);
        float p = __expf(s - mn);
        l = l * corr + p;
        const float4* vrow = (const float4*)&Vls[r][0];
        #pragma unroll
        for (int d4 = 0; d4 < 8; ++d4) {
          float4 vv = vrow[d4];
          o[4*d4 + 0] = o[4*d4 + 0] * corr + p * vv.x;
          o[4*d4 + 1] = o[4*d4 + 1] * corr + p * vv.y;
          o[4*d4 + 2] = o[4*d4 + 2] * corr + p * vv.z;
          o[4*d4 + 3] = o[4*d4 + 3] * corr + p * vv.w;
        }
        m = mn;
      }
    }
    __syncthreads();
  }
  if (active) {
    float inv = 1.f / l;
    float* ob = &ctx[((size_t)(b * NQL + q)) * DM + h * DHD];
    #pragma unroll
    for (int d4 = 0; d4 < 8; ++d4) {
      float4 ov;
      ov.x = o[4*d4+0] * inv; ov.y = o[4*d4+1] * inv;
      ov.z = o[4*d4+2] * inv; ov.w = o[4*d4+3] * inv;
      *(float4*)&ob[4*d4] = ov;
    }
  }
}

// ---------------------------------------------------------------------------
// LayerNorm(x + res) * g + b, one block per row, D = 256 or 512.
// wave32 shuffle reduction + 8-entry shared combine.
// ---------------------------------------------------------------------------
__global__ __launch_bounds__(256) void ln_residual_kernel(
    const float* __restrict__ x, const float* __restrict__ r,
    const float* __restrict__ g, const float* __restrict__ bt,
    float* __restrict__ y, int D)
{
  __shared__ float part[8];
  const int row = blockIdx.x, tid = threadIdx.x;
  const int ept = D >> 8;
  float vals[2] = {0.f, 0.f};
  float s = 0.f;
  for (int e = 0; e < ept; ++e) {
    int c = tid + e * 256;
    float v = x[(size_t)row * D + c] + r[(size_t)row * D + c];
    vals[e] = v; s += v;
  }
  s = wave_reduce_add(s);
  if ((tid & 31) == 0) part[tid >> 5] = s;
  __syncthreads();
  float tot = 0.f;
  #pragma unroll
  for (int w = 0; w < 8; ++w) tot += part[w];
  float mean = tot / (float)D;
  __syncthreads();
  float sq = 0.f;
  for (int e = 0; e < ept; ++e) { float d = vals[e] - mean; sq += d * d; }
  sq = wave_reduce_add(sq);
  if ((tid & 31) == 0) part[tid >> 5] = sq;
  __syncthreads();
  float tot2 = 0.f;
  #pragma unroll
  for (int w = 0; w < 8; ++w) tot2 += part[w];
  float rstd = rsqrtf(tot2 / (float)D + 1e-5f);
  for (int e = 0; e < ept; ++e) {
    int c = tid + e * 256;
    y[(size_t)row * D + c] = (vals[e] - mean) * rstd * g[c] + bt[c];
  }
}

// LayerNorm over concat([a,b], -1) with D=512, one block per row
__global__ __launch_bounds__(256) void concat_ln_kernel(
    const float* __restrict__ a, const float* __restrict__ b2,
    const float* __restrict__ g, const float* __restrict__ bt,
    float* __restrict__ y)
{
  __shared__ float part[8];
  const int row = blockIdx.x, tid = threadIdx.x;
  float v0 = a [(size_t)row * DM + tid];
  float v1 = b2[(size_t)row * DM + tid];
  float s = wave_reduce_add(v0 + v1);
  if ((tid & 31) == 0) part[tid >> 5] = s;
  __syncthreads();
  float tot = 0.f;
  #pragma unroll
  for (int w = 0; w < 8; ++w) tot += part[w];
  float mean = tot / 512.f;
  __syncthreads();
  float d0 = v0 - mean, d1 = v1 - mean;
  float sq = wave_reduce_add(d0 * d0 + d1 * d1);
  if ((tid & 31) == 0) part[tid >> 5] = sq;
  __syncthreads();
  float tot2 = 0.f;
  #pragma unroll
  for (int w = 0; w < 8; ++w) tot2 += part[w];
  float rstd = rsqrtf(tot2 / 512.f + 1e-5f);
  y[(size_t)row * 512 + tid]       = d0 * rstd * g[tid]       + bt[tid];
  y[(size_t)row * 512 + tid + 256] = d1 * rstd * g[tid + 256] + bt[tid + 256];
}

// ---------------------------------------------------------------------------
// softmax over L*P=16 per (b,q,h); writes identical result to both out slices
// ---------------------------------------------------------------------------
__global__ void softmax16_kernel(const float* __restrict__ awr,
                                 float* __restrict__ o1, float* __restrict__ o2)
{
  int gid = blockIdx.x * 256 + threadIdx.x;
  if (gid >= BQ * NH) return;
  const float* p = awr + (size_t)gid * 16;
  float v[16], mx = -1e30f;
  #pragma unroll
  for (int j = 0; j < 16; ++j) { v[j] = p[j]; mx = fmaxf(mx, v[j]); }
  float s = 0.f;
  #pragma unroll
  for (int j = 0; j < 16; ++j) { v[j] = __expf(v[j] - mx); s += v[j]; }
  float inv = 1.f / s;
  #pragma unroll
  for (int j = 0; j < 16; ++j) {
    float r = v[j] * inv;
    o1[(size_t)gid * 16 + j] = r;
    o2[(size_t)gid * 16 + j] = r;
  }
}

// loc[b,q,h,l,p] = ref[b,q,l] + off[b,q,h,l,p] / norm[l]
__global__ void loc_kernel(const float* __restrict__ ref, const float* __restrict__ offr,
                           float* __restrict__ out_loc,
                           float n0, float n1, float n2, float n3)
{
  int idx = blockIdx.x * 256 + threadIdx.x;
  if (idx >= BQ * NH * NLV * NPP) return;
  int l  = (idx >> 2) & 3;
  int bq = idx >> 7;
  float norm = (l == 0) ? n0 : (l == 1) ? n1 : (l == 2) ? n2 : n3;
  out_loc[idx] = ref[bq * NLV + l] + offr[idx] / norm;
}

// ---------------------------------------------------------------------------
// MSDA sampling: one wave per (b,q,h), lane = channel d. 16 linear samples
// from bf16 value tensor, weighted by softmaxed attention weights.
// ---------------------------------------------------------------------------
__global__ __launch_bounds__(256) void msda_gather_kernel(
    const float* __restrict__ loc, const float* __restrict__ aw,
    const unsigned short* __restrict__ val, float* __restrict__ ctx,
    int T, int t0, int t1, int t2, int t3)
{
  const int wid  = blockIdx.x * 8 + (threadIdx.x >> 5);
  const int lane = threadIdx.x & 31;
  const int h  = wid & 7;
  const int bq = wid >> 3;
  const int b  = bq / NQL;
  const int sizes[4]  = {t0, t1, t2, t3};
  const int starts[4] = {0, t0, t0 + t1, t0 + t1 + t2};
  const size_t vbase = ((size_t)b * T) * DM + h * DHD + lane;
  const int lbase = bq * 128 + h * 16;
  float acc = 0.f;
  #pragma unroll
  for (int l = 0; l < 4; ++l) {
    const int Tl = sizes[l], st = starts[l];
    #pragma unroll
    for (int p = 0; p < 4; ++p) {
      int idx = lbase + l * 4 + p;
      float x  = loc[idx] * (float)Tl - 0.5f;
      float xf = floorf(x);
      float w  = x - xf;
      int x0 = (int)xf;
      float g0 = 0.f, g1 = 0.f;
      if (x0 >= 0 && x0 < Tl)         g0 = bf2f(val[vbase + (size_t)(st + x0) * DM]);
      if (x0 + 1 >= 0 && x0 + 1 < Tl) g1 = bf2f(val[vbase + (size_t)(st + x0 + 1) * DM]);
      acc += aw[idx] * (g0 * (1.f - w) + g1 * w);
    }
  }
  ctx[(size_t)bq * DM + h * DHD + lane] = acc;
}

// ===========================================================================
extern "C" void kernel_launch(void* const* d_in, const int* in_sizes, int n_in,
                              void* d_out, int out_size, void* d_ws, size_t ws_size,
                              hipStream_t stream) {
  (void)in_sizes; (void)n_in; (void)out_size; (void)ws_size;

  const float* tgt   = (const float*)d_in[0];
  const float* qpos  = (const float*)d_in[1];
  const float* refv  = (const float*)d_in[2];
  const float* refa  = (const float*)d_in[3];
  const unsigned char* qmask = (const unsigned char*)d_in[4];
  const float* vsrc  = (const float*)d_in[5];
  const float* asrc  = (const float*)d_in[6];
  const unsigned char* vpad = (const unsigned char*)d_in[7];
  const unsigned char* apad = (const unsigned char*)d_in[8];
  const float* sa_q_w = (const float*)d_in[9];   const float* sa_q_b = (const float*)d_in[10];
  const float* sa_k_w = (const float*)d_in[11];  const float* sa_k_b = (const float*)d_in[12];
  const float* sa_v_w = (const float*)d_in[13];  const float* sa_v_b = (const float*)d_in[14];
  const float* sa_o_w = (const float*)d_in[15];  const float* sa_o_b = (const float*)d_in[16];
  const float* off_w  = (const float*)d_in[17];  const float* off_b  = (const float*)d_in[18];
  const float* aw_w   = (const float*)d_in[19];  const float* aw_b   = (const float*)d_in[20];
  const float* vproj_w= (const float*)d_in[21];  const float* vproj_b= (const float*)d_in[22];
  const float* oproj_w= (const float*)d_in[23];  const float* oproj_b= (const float*)d_in[24];
  const float* lin1_w = (const float*)d_in[25];  const float* lin1_b = (const float*)d_in[26];
  const float* lin2_w = (const float*)d_in[27];  const float* lin2_b = (const float*)d_in[28];
  const float* lin3_w = (const float*)d_in[29];  const float* lin3_b = (const float*)d_in[30];
  const float* n1_g = (const float*)d_in[31];    const float* n1_b = (const float*)d_in[32];
  const float* n2_g = (const float*)d_in[33];    const float* n2_b = (const float*)d_in[34];
  const float* n3_g = (const float*)d_in[35];    const float* n3_b = (const float*)d_in[36];
  const float* n4_g = (const float*)d_in[37];    const float* n4_b = (const float*)d_in[38];

  // Output slices (tuple return, flattened in order)
  const size_t NBQD = (size_t)BQ * DM;              // 2,457,600
  const size_t NLOC = (size_t)BQ * NH * NLV * NPP;  // 1,228,800
  float* o_out  = (float*)d_out;
  float* o_tv   = o_out  + NBQD;
  float* o_ta   = o_tv   + NBQD;
  float* o_locv = o_ta   + NBQD;
  float* o_awv  = o_locv + NLOC;
  float* o_loca = o_awv  + NLOC;
  float* o_awa  = o_loca + NLOC;

  // Workspace allocator
  size_t cur = 0;
  auto alloc = [&](size_t bytes) -> void* {
    void* p = (char*)d_ws + cur;
    cur += (bytes + 255) & ~(size_t)255;
    return p;
  };
  float* qbuf  = (float*)alloc(NBQD * 4);
  float* Qp    = (float*)alloc(NBQD * 4);
  float* Kp    = (float*)alloc(NBQD * 4);
  float* Vp    = (float*)alloc(NBQD * 4);
  float* actx  = (float*)alloc(NBQD * 4);
  float* sa    = (float*)alloc(NBQD * 4);
  float* tgt1  = (float*)alloc(NBQD * 4);
  float* qe    = (float*)alloc(NBQD * 4);
  float* offr  = (float*)alloc(NLOC * 4);
  float* awr   = (float*)alloc(NLOC * 4);
  float* ctxv  = (float*)alloc(NBQD * 4);
  float* ctxa  = (float*)alloc(NBQD * 4);
  float* tv    = (float*)alloc(NBQD * 4);
  float* ta    = (float*)alloc(NBQD * 4);
  float* catln = (float*)alloc((size_t)BQ * 512 * 4);
  float* fused = (float*)alloc(NBQD * 4);
  float* h1    = (float*)alloc((size_t)BQ * DFF * 4);
  float* ffn   = (float*)alloc(NBQD * 4);
  unsigned short* valv = (unsigned short*)alloc((size_t)BB * TVID * DM * 2);
  unsigned short* vala = (unsigned short*)alloc((size_t)BB * TAUD * DM * 2);

  const dim3 blk(256);
  const int MV = BB * TVID;   // 245760
  const int MA = BB * TAUD;   // 92160

  // 1. q = tgt + query_pos
  add_kernel<<<dim3((unsigned)(NBQD / 256)), blk, 0, stream>>>(tgt, qpos, qbuf, (int)NBQD);

  // 2. self-attention projections (q,q,tgt)
  gemm_wmma<0,0,0><<<dim3(DM/64, BQ/64), blk, 0, stream>>>(qbuf, sa_q_w, sa_q_b, Qp, nullptr, nullptr, BQ, DM, DM);
  gemm_wmma<0,0,0><<<dim3(DM/64, BQ/64), blk, 0, stream>>>(qbuf, sa_k_w, sa_k_b, Kp, nullptr, nullptr, BQ, DM, DM);
  gemm_wmma<0,0,0><<<dim3(DM/64, BQ/64), blk, 0, stream>>>(tgt,  sa_v_w, sa_v_b, Vp, nullptr, nullptr, BQ, DM, DM);

  // 3. attention + output projection
  attn_kernel<<<dim3(BB * NH * 2), blk, 0, stream>>>(Qp, Kp, Vp, qmask, actx);
  gemm_wmma<0,0,0><<<dim3(DM/64, BQ/64), blk, 0, stream>>>(actx, sa_o_w, sa_o_b, sa, nullptr, nullptr, BQ, DM, DM);

  // 4. tgt1 = LN(tgt + sa; n2); qe = tgt1 + qpos
  ln_residual_kernel<<<dim3(BQ), blk, 0, stream>>>(tgt, sa, n2_g, n2_b, tgt1, DM);
  add_kernel<<<dim3((unsigned)(NBQD / 256)), blk, 0, stream>>>(tgt1, qpos, qe, (int)NBQD);

  // 5. sampling offsets & attention weights (shared by both modalities)
  gemm_wmma<0,0,0><<<dim3(128/64, BQ/64), blk, 0, stream>>>(qe, off_w, off_b, offr, nullptr, nullptr, BQ, 128, DM);
  gemm_wmma<0,0,0><<<dim3(128/64, BQ/64), blk, 0, stream>>>(qe, aw_w,  aw_b,  awr,  nullptr, nullptr, BQ, 128, DM);
  softmax16_kernel<<<dim3((BQ * NH) / 256), blk, 0, stream>>>(awr, o_awv, o_awa);
  loc_kernel<<<dim3((unsigned)(NLOC / 256)), blk, 0, stream>>>(refv, offr, o_locv, 4096.f, 2048.f, 1024.f, 512.f);
  loc_kernel<<<dim3((unsigned)(NLOC / 256)), blk, 0, stream>>>(refa, offr, o_loca, 1536.f,  768.f,  384.f, 192.f);

  // 6. value projections (bf16 out, pad rows zeroed)
  gemm_wmma<0,1,1><<<dim3(DM/64, MV/64), blk, 0, stream>>>(vsrc, vproj_w, vproj_b, nullptr, valv, vpad, MV, DM, DM);
  gemm_wmma<0,1,1><<<dim3(DM/64, MA/64), blk, 0, stream>>>(asrc, vproj_w, vproj_b, nullptr, vala, apad, MA, DM, DM);

  // 7. video branch: gather -> oproj -> LN(n1)
  msda_gather_kernel<<<dim3(BQ), blk, 0, stream>>>(o_locv, o_awv, valv, ctxv, TVID, 4096, 2048, 1024, 512);
  gemm_wmma<0,0,0><<<dim3(DM/64, BQ/64), blk, 0, stream>>>(ctxv, oproj_w, oproj_b, tv, nullptr, nullptr, BQ, DM, DM);
  ln_residual_kernel<<<dim3(BQ), blk, 0, stream>>>(tgt1, tv, n1_g, n1_b, o_tv, DM);

  // 8. audio branch
  msda_gather_kernel<<<dim3(BQ), blk, 0, stream>>>(o_loca, o_awa, vala, ctxa, TAUD, 1536, 768, 384, 192);
  gemm_wmma<0,0,0><<<dim3(DM/64, BQ/64), blk, 0, stream>>>(ctxa, oproj_w, oproj_b, ta, nullptr, nullptr, BQ, DM, DM);
  ln_residual_kernel<<<dim3(BQ), blk, 0, stream>>>(tgt1, ta, n1_g, n1_b, o_ta, DM);

  // 9. fusion + FFN
  concat_ln_kernel<<<dim3(BQ), blk, 0, stream>>>(o_tv, o_ta, n4_g, n4_b, catln);
  gemm_wmma<0,0,0><<<dim3(DM/64,  BQ/64), blk, 0, stream>>>(catln, lin3_w, lin3_b, fused, nullptr, nullptr, BQ, DM, 512);
  gemm_wmma<1,0,0><<<dim3(DFF/64, BQ/64), blk, 0, stream>>>(fused, lin1_w, lin1_b, h1,    nullptr, nullptr, BQ, DFF, DM);
  gemm_wmma<0,0,0><<<dim3(DM/64,  BQ/64), blk, 0, stream>>>(h1,    lin2_w, lin2_b, ffn,   nullptr, nullptr, BQ, DM, DFF);
  ln_residual_kernel<<<dim3(BQ), blk, 0, stream>>>(fused, ffn, n3_g, n3_b, o_out, DM);
}
